// FrequencyEnhancedMambaBlock_71906342470082
// MI455X (gfx1250) — compile-verified, matrix operand order checked by reading, not run on hardware
//
#include <hip/hip_runtime.h>
#include <hip/hip_bf16.h>
#include <math.h>

#define B_    8
#define S_    1024
#define D_    768
#define DI_   1536
#define DS_   16
#define DTR_  48
#define DTRP_ 64      // dt_rank padded to K multiple of 32
#define NPX_  128     // x_proj out (80) padded to N multiple of 64
#define NBHR_ 62      // HR bins: k = 24..85
#define NBBR_ 8       // BR bins: k = 6..13
#define NB_   70
#define ROWS_ (B_*S_) // 8192

typedef __attribute__((ext_vector_type(16))) __bf16 bf16x16;
typedef __attribute__((ext_vector_type(8)))  float  f32x8;

union BPack { uint4 u[2]; bf16x16 v; };

__device__ __forceinline__ bf16x16 load_frag(const __bf16* p) {
  BPack pk;
  pk.u[0] = *reinterpret_cast<const uint4*>(p);       // K = q+0..q+7
  pk.u[1] = *reinterpret_cast<const uint4*>(p + 16);  // K = q+16..q+23
  return pk.v;
}

// ---------------------------------------------------------------------------
// Generic NT GEMM: out[M,N] = act(A[M,K] * W[N,K]^T + bias) (+resid), bf16 in,
// f32 accumulate via v_wmma_f32_16x16x32_bf16. One wave -> 16x64 tile.
// Requires M%16==0, N%64==0, K%32==0.
// act: 0=none, 1=silu, 2=softplus
// ---------------------------------------------------------------------------
__global__ __launch_bounds__(256)
void gemm_bf16_nt(const __bf16* __restrict__ A, const __bf16* __restrict__ W,
                  const float* __restrict__ bias,
                  float* __restrict__ outF, int ldF,
                  __bf16* __restrict__ outB, int ldB, int colOfsB,
                  const float* __restrict__ resid, int ldR,
                  int M, int N, int K, int act)
{
  const int wid = blockIdx.x * 8 + (threadIdx.x >> 5);
  const int Mt = M >> 4;
  const int Ng = N >> 6;
  if (wid >= Mt * Ng) return;                 // whole-wave exit: EXEC stays all-1s
  const int mt = wid % Mt;
  const int nb = (wid / Mt) << 6;
  const int lane  = threadIdx.x & 31;
  const int lrow  = lane & 15;
  const int lhalf = lane >> 4;

  const __bf16* Ap = A + (size_t)(mt * 16 + lrow) * K + lhalf * 8;
  const __bf16* Wp = W + (size_t)(nb + lrow) * K + lhalf * 8;
  const size_t wstride = (size_t)16 * K;

  f32x8 acc0 = {0.f,0.f,0.f,0.f,0.f,0.f,0.f,0.f};
  f32x8 acc1 = acc0, acc2 = acc0, acc3 = acc0;

  for (int k = 0; k < K; k += 32) {
    __builtin_prefetch(Ap + k + 256, 0, 1);
    __builtin_prefetch(Wp + k + 256, 0, 1);
    bf16x16 a  = load_frag(Ap + k);
    bf16x16 b0 = load_frag(Wp + k);
    bf16x16 b1 = load_frag(Wp + wstride + k);
    bf16x16 b2 = load_frag(Wp + 2 * wstride + k);
    bf16x16 b3 = load_frag(Wp + 3 * wstride + k);
    acc0 = __builtin_amdgcn_wmma_f32_16x16x32_bf16(false, a, false, b0, (short)0, acc0, false, false);
    acc1 = __builtin_amdgcn_wmma_f32_16x16x32_bf16(false, a, false, b1, (short)0, acc1, false, false);
    acc2 = __builtin_amdgcn_wmma_f32_16x16x32_bf16(false, a, false, b2, (short)0, acc2, false, false);
    acc3 = __builtin_amdgcn_wmma_f32_16x16x32_bf16(false, a, false, b3, (short)0, acc3, false, false);
  }

  #pragma unroll
  for (int nt = 0; nt < 4; ++nt) {
    const int col = nb + nt * 16 + lrow;
    const float bv = bias ? bias[col] : 0.f;
    f32x8 acc = (nt == 0) ? acc0 : (nt == 1) ? acc1 : (nt == 2) ? acc2 : acc3;
    #pragma unroll
    for (int r = 0; r < 8; ++r) {
      const int row = mt * 16 + r + 8 * lhalf;
      float v = acc[r] + bv;
      if (act == 1)      v = v / (1.f + expf(-v));               // silu
      else if (act == 2) v = (v > 20.f) ? v : log1pf(expf(v));   // softplus
      if (resid) v += resid[(size_t)row * ldR + col];
      if (outF)  outF[(size_t)row * ldF + col] = v;
      if (outB)  outB[(size_t)row * ldB + colOfsB + col] = (__bf16)v;
    }
  }
}

// ---------------------------------------------------------------------------
// f32 -> bf16 weight conversion with optional zero padding.
// ---------------------------------------------------------------------------
__global__ __launch_bounds__(256)
void convert_pad(const float* __restrict__ src, __bf16* __restrict__ dst,
                 int rows, int cols, int drows, int dcols)
{
  const int idx = blockIdx.x * 256 + threadIdx.x;
  if (idx >= drows * dcols) return;
  const int r = idx / dcols, c = idx % dcols;
  float v = (r < rows && c < cols) ? src[(size_t)r * cols + c] : 0.f;
  dst[idx] = (__bf16)v;
}

// ---------------------------------------------------------------------------
// LayerNorm over D=768, one block per (b,s) row. Writes f32 + bf16.
// ---------------------------------------------------------------------------
__global__ __launch_bounds__(256)
void ln_kernel(const float* __restrict__ x, const float* __restrict__ w,
               const float* __restrict__ bb, float* __restrict__ z,
               __bf16* __restrict__ zb)
{
  const size_t row = blockIdx.x;
  const float* xr = x + row * D_;
  __shared__ float red[256];
  const int tid = threadIdx.x;

  float s0 = 0.f;
  for (int i = tid; i < D_; i += 256) s0 += xr[i];
  red[tid] = s0; __syncthreads();
  for (int st = 128; st > 0; st >>= 1) { if (tid < st) red[tid] += red[tid + st]; __syncthreads(); }
  const float mu = red[0] / D_;
  __syncthreads();

  float v0 = 0.f;
  for (int i = tid; i < D_; i += 256) { float dd = xr[i] - mu; v0 += dd * dd; }
  red[tid] = v0; __syncthreads();
  for (int st = 128; st > 0; st >>= 1) { if (tid < st) red[tid] += red[tid + st]; __syncthreads(); }
  const float rstd = rsqrtf(red[0] / D_ + 1e-5f);

  for (int i = tid; i < D_; i += 256) {
    float val = (xr[i] - mu) * rstd * w[i] + bb[i];
    z[row * D_ + i] = val;
    zb[row * D_ + i] = (__bf16)val;
  }
}

// ---------------------------------------------------------------------------
// Depthwise causal conv (width 4) + bias + SiLU. xin = xz[:, :DI].
// ---------------------------------------------------------------------------
__global__ __launch_bounds__(256)
void conv_silu(const float* __restrict__ xz, const float* __restrict__ cw,
               const float* __restrict__ cb, float* __restrict__ xc,
               __bf16* __restrict__ xcb)
{
  const size_t idx = (size_t)blockIdx.x * 256 + threadIdx.x;
  if (idx >= (size_t)ROWS_ * DI_) return;
  const int c = idx % DI_;
  const int s = (idx / DI_) % S_;
  const int b = idx / ((size_t)DI_ * S_);
  float acc = cb[c];
  #pragma unroll
  for (int j = 0; j < 4; ++j) {
    const int ss = s - 3 + j;
    if (ss >= 0)
      acc += cw[c * 4 + j] * xz[((size_t)b * S_ + ss) * (2 * DI_) + c];
  }
  const float v = acc / (1.f + expf(-acc));
  xc[idx] = v;
  xcb[idx] = (__bf16)v;
}

// ---------------------------------------------------------------------------
// Split dbc (ld=128) into dt_r (bf16, K padded to 64), Bm, Cm (f32).
// ---------------------------------------------------------------------------
__global__ __launch_bounds__(256)
void extract_dbc(const float* __restrict__ dbc, __bf16* __restrict__ dtr,
                 float* __restrict__ Bm, float* __restrict__ Cm)
{
  const size_t idx = (size_t)blockIdx.x * 256 + threadIdx.x;
  if (idx >= (size_t)ROWS_ * 96) return;
  const size_t r = idx / 96;
  const int col = idx % 96;
  if (col < DTR_)        dtr[r * DTRP_ + col] = (__bf16)dbc[r * NPX_ + col];
  else if (col < 64)     Bm[r * DS_ + (col - DTR_)] = dbc[r * NPX_ + col];
  else if (col < 80)     Cm[r * DS_ + (col - 64)]   = dbc[r * NPX_ + col];
  else                   dtr[r * DTRP_ + DTR_ + (col - 80)] = (__bf16)0.f;  // zero pad
}

// ---------------------------------------------------------------------------
// Selective scan: one thread per (b,d) channel, 16 states in registers.
// Fuses y = (scan + D*xc) * silu(gate) and casts to bf16 for out_proj.
// ---------------------------------------------------------------------------
__global__ __launch_bounds__(256)
void scan_kernel(const float* __restrict__ dt, const float* __restrict__ xc,
                 const float* __restrict__ Bm, const float* __restrict__ Cm,
                 const float* __restrict__ xz, const float* __restrict__ A_log,
                 const float* __restrict__ Dp, __bf16* __restrict__ ybf)
{
  const int b = blockIdx.x / (DI_ / 256);
  const int d = (blockIdx.x % (DI_ / 256)) * 256 + threadIdx.x;
  float a[DS_], h[DS_];
  #pragma unroll
  for (int n = 0; n < DS_; ++n) { a[n] = -expf(A_log[(size_t)d * DS_ + n]); h[n] = 0.f; }
  const float dpar = Dp[d];
  __shared__ float sB[DS_], sC[DS_];

  for (int s = 0; s < S_; ++s) {
    const size_t row = (size_t)b * S_ + s;
    if (threadIdx.x < DS_)            sB[threadIdx.x]       = Bm[row * DS_ + threadIdx.x];
    else if (threadIdx.x < 2 * DS_)   sC[threadIdx.x - DS_] = Cm[row * DS_ + threadIdx.x - DS_];
    __syncthreads();
    const float dtv = dt[row * DI_ + d];
    const float xv  = xc[row * DI_ + d];
    const float dx  = dtv * xv;
    float y = 0.f;
    #pragma unroll
    for (int n = 0; n < DS_; ++n) {
      h[n] = expf(dtv * a[n]) * h[n] + dx * sB[n];
      y = fmaf(h[n], sC[n], y);
    }
    const float g = xz[row * (2 * DI_) + DI_ + d];
    const float out = (y + dpar * xv) * (g / (1.f + expf(-g)));
    ybf[row * DI_ + d] = (__bf16)out;
    __syncthreads();
  }
}

// ---------------------------------------------------------------------------
// Band DFT basis: cos/sin for the 70 kept rfft bins (HR 24..85, BR 6..13).
// ---------------------------------------------------------------------------
__global__ __launch_bounds__(256)
void basis_init(float* __restrict__ bc, float* __restrict__ bs)
{
  const int idx = blockIdx.x * 256 + threadIdx.x;
  if (idx >= NB_ * S_) return;
  const int j = idx >> 10, t = idx & (S_ - 1);
  const int k = (j < NBHR_) ? (24 + j) : (6 + (j - NBHR_));
  const int m = (k * t) & (S_ - 1);                    // exact mod-1024 reduction
  const float ang = (float)(6.283185307179586 * (double)m / (double)S_);
  bc[idx] = cosf(ang);
  bs[idx] = sinf(ang);
}

// Forward projection: Fc/Fs[b,j,d] = sum_t basis[j,t] * z[b,t,d].
// grid = (jchunk=5, dchunk=3, b=8); each chunk handles 14 bins.
__global__ __launch_bounds__(256)
void dft_fwd(const float* __restrict__ z, const float* __restrict__ bc,
             const float* __restrict__ bs, float* __restrict__ Fc,
             float* __restrict__ Fs)
{
  const int b = blockIdx.z;
  const int d = blockIdx.y * 256 + threadIdx.x;
  const int j0 = blockIdx.x * 14;
  float ac[14], as_[14];
  #pragma unroll
  for (int jj = 0; jj < 14; ++jj) { ac[jj] = 0.f; as_[jj] = 0.f; }
  const float* zb = z + (size_t)b * S_ * D_ + d;
  for (int t = 0; t < S_; ++t) {
    const float zv = zb[(size_t)t * D_];
    #pragma unroll
    for (int jj = 0; jj < 14; ++jj) {
      ac[jj]  = fmaf(bc[(j0 + jj) * S_ + t], zv, ac[jj]);
      as_[jj] = fmaf(bs[(j0 + jj) * S_ + t], zv, as_[jj]);
    }
  }
  #pragma unroll
  for (int jj = 0; jj < 14; ++jj) {
    Fc[((size_t)b * NB_ + j0 + jj) * D_ + d] = ac[jj];
    Fs[((size_t)b * NB_ + j0 + jj) * D_ + d] = as_[jj];
  }
}

// Inverse band synthesis: writes hr_enh, br_enh outputs and bf16 "combined"
// into cat columns [768,1536). grid = (d=768, b=8).
__global__ __launch_bounds__(256)
void dft_inv(const float* __restrict__ Fc, const float* __restrict__ Fs,
             const float* __restrict__ bc, const float* __restrict__ bs,
             const float* __restrict__ ghr, const float* __restrict__ gbr,
             float* __restrict__ hr_out, float* __restrict__ br_out,
             __bf16* __restrict__ cat)
{
  const int d = blockIdx.x, b = blockIdx.y;
  __shared__ float sc[NB_], ss[NB_];
  const int tid = threadIdx.x;
  if (tid < NB_) {
    sc[tid] = Fc[((size_t)b * NB_ + tid) * D_ + d];
    ss[tid] = Fs[((size_t)b * NB_ + tid) * D_ + d];
  }
  __syncthreads();
  const float gh = ghr[d] * (2.f / S_);
  const float gb = gbr[d] * (2.f / S_);
  for (int t = tid; t < S_; t += 256) {
    float hr = 0.f, br = 0.f;
    #pragma unroll 2
    for (int j = 0; j < NBHR_; ++j)
      hr += sc[j] * bc[j * S_ + t] + ss[j] * bs[j * S_ + t];
    #pragma unroll
    for (int j = NBHR_; j < NB_; ++j)
      br += sc[j] * bc[j * S_ + t] + ss[j] * bs[j * S_ + t];
    const float hrv = hr * gh, brv = br * gb;
    const size_t row = (size_t)b * S_ + t;
    hr_out[row * D_ + d] = hrv;
    br_out[row * D_ + d] = brv;
    cat[row * (2 * D_) + D_ + d] = (__bf16)(hrv + brv);
  }
}

// ---------------------------------------------------------------------------
// Host orchestration
// ---------------------------------------------------------------------------
extern "C" void kernel_launch(void* const* d_in, const int* in_sizes, int n_in,
                              void* d_out, int out_size, void* d_ws, size_t ws_size,
                              hipStream_t stream)
{
  const float* x          = (const float*)d_in[0];
  const float* ln_w       = (const float*)d_in[1];
  const float* ln_b       = (const float*)d_in[2];
  const float* in_proj_w  = (const float*)d_in[3];
  const float* in_proj_b  = (const float*)d_in[4];
  const float* conv_w     = (const float*)d_in[5];
  const float* conv_b     = (const float*)d_in[6];
  const float* x_proj_w   = (const float*)d_in[7];
  const float* dt_proj_w  = (const float*)d_in[8];
  const float* dt_proj_b  = (const float*)d_in[9];
  const float* A_log      = (const float*)d_in[10];
  const float* D_param    = (const float*)d_in[11];
  const float* out_proj_w = (const float*)d_in[12];
  const float* out_proj_b = (const float*)d_in[13];
  const float* g_hr       = (const float*)d_in[14];
  const float* g_br       = (const float*)d_in[15];
  const float* fuse1_w    = (const float*)d_in[16];
  const float* fuse1_b    = (const float*)d_in[17];
  const float* fuse2_w    = (const float*)d_in[18];
  const float* fuse2_b    = (const float*)d_in[19];

  float* out0 = (float*)d_out;                       // output  (B,S,D)
  float* hr_o = out0 + (size_t)ROWS_ * D_;           // hr_enh
  float* br_o = hr_o + (size_t)ROWS_ * D_;           // br_enh

  // ---- workspace bump allocator (256B aligned) ----
  char* base = (char*)d_ws;
  size_t off = 0;
  auto alloc = [&](size_t bytes) -> void* {
    off = (off + 255) & ~(size_t)255;
    void* p = base + off;
    off += bytes;
    return p;
  };
  float*  z      = (float*)alloc((size_t)ROWS_ * D_ * 4);
  __bf16* z_bf   = (__bf16*)alloc((size_t)ROWS_ * D_ * 2);
  float*  xz     = (float*)alloc((size_t)ROWS_ * 2 * DI_ * 4);
  float*  xc     = (float*)alloc((size_t)ROWS_ * DI_ * 4);
  __bf16* xc_bf  = (__bf16*)alloc((size_t)ROWS_ * DI_ * 2);
  float*  dbc    = (float*)alloc((size_t)ROWS_ * NPX_ * 4);
  __bf16* dtr_bf = (__bf16*)alloc((size_t)ROWS_ * DTRP_ * 2);
  float*  Bm     = (float*)alloc((size_t)ROWS_ * DS_ * 4);
  float*  Cm     = (float*)alloc((size_t)ROWS_ * DS_ * 4);
  float*  dtv    = (float*)alloc((size_t)ROWS_ * DI_ * 4);
  __bf16* y_bf   = (__bf16*)alloc((size_t)ROWS_ * DI_ * 2);
  __bf16* cat_bf = (__bf16*)alloc((size_t)ROWS_ * 2 * D_ * 2);
  __bf16* f_bf   = (__bf16*)alloc((size_t)ROWS_ * D_ * 2);
  float*  Fc     = (float*)alloc((size_t)B_ * NB_ * D_ * 4);
  float*  Fs     = (float*)alloc((size_t)B_ * NB_ * D_ * 4);
  float*  basC   = (float*)alloc((size_t)NB_ * S_ * 4);
  float*  basS   = (float*)alloc((size_t)NB_ * S_ * 4);
  __bf16* inw    = (__bf16*)alloc((size_t)2 * DI_ * D_ * 2);
  __bf16* xpw    = (__bf16*)alloc((size_t)NPX_ * DI_ * 2);
  __bf16* dtw    = (__bf16*)alloc((size_t)DI_ * DTRP_ * 2);
  __bf16* outw   = (__bf16*)alloc((size_t)D_ * DI_ * 2);
  __bf16* f1w    = (__bf16*)alloc((size_t)D_ * 2 * DI_ * 2);  // fuse1_w is (D, 2D) = 768x1536
  __bf16* f2w    = (__bf16*)alloc((size_t)D_ * D_ * 2);
  (void)ws_size; (void)n_in; (void)in_sizes; (void)out_size;

  auto cblk = [](size_t n) { return (int)((n + 255) / 256); };

  // 1) weight conversions (+padding) and DFT basis
  convert_pad<<<cblk((size_t)2*DI_*D_), 256, 0, stream>>>(in_proj_w, inw, 2*DI_, D_, 2*DI_, D_);
  convert_pad<<<cblk((size_t)NPX_*DI_), 256, 0, stream>>>(x_proj_w, xpw, 80, DI_, NPX_, DI_);
  convert_pad<<<cblk((size_t)DI_*DTRP_), 256, 0, stream>>>(dt_proj_w, dtw, DI_, DTR_, DI_, DTRP_);
  convert_pad<<<cblk((size_t)D_*DI_), 256, 0, stream>>>(out_proj_w, outw, D_, DI_, D_, DI_);
  convert_pad<<<cblk((size_t)D_*2*DI_), 256, 0, stream>>>(fuse1_w, f1w, D_, 2*D_, D_, 2*D_);
  convert_pad<<<cblk((size_t)D_*D_), 256, 0, stream>>>(fuse2_w, f2w, D_, D_, D_, D_);
  basis_init<<<cblk((size_t)NB_*S_), 256, 0, stream>>>(basC, basS);

  // 2) LayerNorm
  ln_kernel<<<ROWS_, 256, 0, stream>>>(x, ln_w, ln_b, z, z_bf);

  // 3) in_proj GEMM: (8192x768)x(3072x768)^T -> xz
  {
    int Mt = ROWS_ / 16, Ng = (2 * DI_) / 64;
    gemm_bf16_nt<<<(Mt * Ng + 7) / 8, 256, 0, stream>>>(
        z_bf, inw, in_proj_b, xz, 2 * DI_, nullptr, 0, 0, nullptr, 0,
        ROWS_, 2 * DI_, D_, 0);
  }

  // 4) depthwise conv + silu
  conv_silu<<<cblk((size_t)ROWS_ * DI_), 256, 0, stream>>>(xz, conv_w, conv_b, xc, xc_bf);

  // 5) x_proj GEMM (N padded to 128, no bias)
  {
    int Mt = ROWS_ / 16, Ng = NPX_ / 64;
    gemm_bf16_nt<<<(Mt * Ng + 7) / 8, 256, 0, stream>>>(
        xc_bf, xpw, nullptr, dbc, NPX_, nullptr, 0, 0, nullptr, 0,
        ROWS_, NPX_, DI_, 0);
  }

  // 6) split dbc -> dt_r(bf16, padded), Bm, Cm
  extract_dbc<<<cblk((size_t)ROWS_ * 96), 256, 0, stream>>>(dbc, dtr_bf, Bm, Cm);

  // 7) dt GEMM + softplus
  {
    int Mt = ROWS_ / 16, Ng = DI_ / 64;
    gemm_bf16_nt<<<(Mt * Ng + 7) / 8, 256, 0, stream>>>(
        dtr_bf, dtw, dt_proj_b, dtv, DI_, nullptr, 0, 0, nullptr, 0,
        ROWS_, DI_, DTRP_, 2);
  }

  // 8) selective scan (fuses D*xc skip and silu(gate) gating, emits bf16)
  scan_kernel<<<B_ * (DI_ / 256), 256, 0, stream>>>(dtv, xc, Bm, Cm, xz, A_log, D_param, y_bf);

  // 9) out_proj GEMM -> cat[:, 0:768] (bf16 only)
  {
    int Mt = ROWS_ / 16, Ng = D_ / 64;
    gemm_bf16_nt<<<(Mt * Ng + 7) / 8, 256, 0, stream>>>(
        y_bf, outw, out_proj_b, nullptr, 0, cat_bf, 2 * D_, 0, nullptr, 0,
        ROWS_, D_, DI_, 0);
  }

  // 10) band DFT branch: forward projection + band-limited synthesis
  dft_fwd<<<dim3(5, 3, B_), 256, 0, stream>>>(z, basC, basS, Fc, Fs);
  dft_inv<<<dim3(D_, B_), 256, 0, stream>>>(Fc, Fs, basC, basS, g_hr, g_br, hr_o, br_o, cat_bf);

  // 11) fuse1 GEMM + silu -> f (bf16)
  {
    int Mt = ROWS_ / 16, Ng = D_ / 64;
    gemm_bf16_nt<<<(Mt * Ng + 7) / 8, 256, 0, stream>>>(
        cat_bf, f1w, fuse1_b, nullptr, 0, f_bf, D_, 0, nullptr, 0,
        ROWS_, D_, 2 * D_, 1);
  }

  // 12) fuse2 GEMM + residual -> output
  {
    int Mt = ROWS_ / 16, Ng = D_ / 64;
    gemm_bf16_nt<<<(Mt * Ng + 7) / 8, 256, 0, stream>>>(
        f_bf, f2w, fuse2_b, out0, D_, nullptr, 0, 0, x, D_,
        ROWS_, D_, D_, 0);
  }
}
